// MaskedSelfAttention_24893630447827
// MI455X (gfx1250) — matrix operand order repaired and checked by reading, not compiled, not run on hardware
//
#include <hip/hip_runtime.h>
#include <hip/hip_bf16.h>
#include <stdint.h>

// Problem constants (from reference)
#define EMBED  1024
#define HDIM   64
#define SEQ    2048
#define NBATCH 4
#define NROWS  (NBATCH * SEQ)   // 8192

typedef __bf16 bf16_t;
typedef float    v8f   __attribute__((ext_vector_type(8)));
typedef bf16_t   v16bf __attribute__((ext_vector_type(16)));
typedef uint32_t u32x4 __attribute__((ext_vector_type(4)));
typedef int      i32x4 __attribute__((ext_vector_type(4)));
typedef int      i32x8 __attribute__((ext_vector_type(8)));

// ---- TDM availability / arity detection -----------------------------------
#if defined(__has_builtin)
#  if __has_builtin(__builtin_amdgcn_tensor_load_to_lds)
#    define HAVE_TDM 1
#  endif
#endif
#ifndef HAVE_TDM
#  define HAVE_TDM 0
#endif
#if __has_include(<hip/amd_detail/amd_gfx1250_TDM.h>)
#  define TDM_SIX_ARG 1   // therock-10.0 headers -> clang-23 6-arg builtin
#else
#  define TDM_SIX_ARG 0   // ROCm 7.2 clang-22 5-arg builtin
#endif

union Frag16 {            // 16 bf16 = 8 VGPRs; loadable as two uint4
    v16bf    v;
    bf16_t   e[16];
    uint4    q[2];
    uint32_t u[8];
};

__device__ __forceinline__ v8f wmma_bf16(v16bf a, v16bf b, v8f c) {
    return __builtin_amdgcn_wmma_f32_16x16x32_bf16(false, a, false, b,
                                                   (short)0, c, false, false);
}

__device__ __forceinline__ v8f zero8() {
    v8f z = {0.f, 0.f, 0.f, 0.f, 0.f, 0.f, 0.f, 0.f};
    return z;
}

__device__ __forceinline__ uint32_t pack_bf16(bf16_t lo, bf16_t hi) {
    union { bf16_t h; uint16_t u; } a, b;
    a.h = lo; b.h = hi;
    return (uint32_t)a.u | ((uint32_t)b.u << 16);
}

// ---------------------------------------------------------------------------
// Kernel 1: repack Wq/Wk/Wv (fp32 [1024][64]) into bf16 WMMA B-fragments.
// Fragment (w,kt,nt): lane holds 8 dwords; dword v = {W[k][n], W[k+1][n]},
// k = kt*32 + 2*(v%4) + 16*(v/4) + 8*(lane>=16), n = nt*16 + lane%16.
// ---------------------------------------------------------------------------
__global__ void __launch_bounds__(256)
wprep(const float* __restrict__ Wq, const float* __restrict__ Wk,
      const float* __restrict__ Wv, bf16_t* __restrict__ wfrag) {
    int idx = blockIdx.x * 256 + threadIdx.x;      // 3*32*4*32*8 = 98304 dwords
    if (idx >= 3 * 32 * 4 * 32 * 8) return;
    int v    = idx & 7;
    int lane = (idx >> 3) & 31;
    int nt   = (idx >> 8) & 3;
    int kt   = (idx >> 10) & 31;
    int w    = idx >> 15;
    const float* W = (w == 0) ? Wq : (w == 1) ? Wk : Wv;
    int n = nt * 16 + (lane & 15);
    int k = kt * 32 + 2 * (v & 3) + ((v >> 2) << 4) + ((lane >> 4) << 3);
    bf16_t lo = (bf16_t)W[k * HDIM + n];
    bf16_t hi = (bf16_t)W[(k + 1) * HDIM + n];
    size_t base = ((((size_t)w * 32 + kt) * 4 + nt) * 32 + lane) * 16 + 2 * v;
    wfrag[base]     = lo;
    wfrag[base + 1] = hi;
}

// ---------------------------------------------------------------------------
// Kernel 2: QKV projection.  One wave -> 16 rows x 64 cols of one projection.
// 32 K-steps x 4 N-tiles of v_wmma_f32_16x16x32_bf16.  Q scaled by 8
// (reference DIVIDES by HDIM^-0.5).
// ---------------------------------------------------------------------------
__global__ void __launch_bounds__(256)
qkv_proj(const float* __restrict__ x, const bf16_t* __restrict__ wfrag,
         const float* __restrict__ bq, const float* __restrict__ bk,
         const float* __restrict__ bv,
         bf16_t* __restrict__ Q, bf16_t* __restrict__ K, bf16_t* __restrict__ V) {
    int wave = threadIdx.x >> 5;
    int lane = threadIdx.x & 31;
    int task = blockIdx.x * 8 + wave;     // 1536 tasks = 3 proj * 512 row tiles
    int p  = task / 512;                  // 0=Q 1=K 2=V
    int rt = task % 512;

    int n  = lane & 15;
    int h  = lane >> 4;
    int kb = h * 8;
    int rowA = rt * 16 + n;               // A-layout: lane%16 = M row
    const float*  xrow = x + (size_t)rowA * EMBED;
    const bf16_t* wf   = wfrag + (size_t)p * 32 * 4 * 32 * 16;

    v8f acc[4];
#pragma unroll
    for (int t = 0; t < 4; ++t) acc[t] = zero8();

    for (int kt = 0; kt < 32; ++kt) {
        Frag16 a;
        const float* xp = xrow + kt * 32 + kb;
        float4 f0 = *(const float4*)(xp);
        float4 f1 = *(const float4*)(xp + 4);
        float4 f2 = *(const float4*)(xp + 16);
        float4 f3 = *(const float4*)(xp + 20);
        a.e[0]=(bf16_t)f0.x; a.e[1]=(bf16_t)f0.y; a.e[2]=(bf16_t)f0.z; a.e[3]=(bf16_t)f0.w;
        a.e[4]=(bf16_t)f1.x; a.e[5]=(bf16_t)f1.y; a.e[6]=(bf16_t)f1.z; a.e[7]=(bf16_t)f1.w;
        a.e[8]=(bf16_t)f2.x; a.e[9]=(bf16_t)f2.y; a.e[10]=(bf16_t)f2.z; a.e[11]=(bf16_t)f2.w;
        a.e[12]=(bf16_t)f3.x; a.e[13]=(bf16_t)f3.y; a.e[14]=(bf16_t)f3.z; a.e[15]=(bf16_t)f3.w;
#pragma unroll
        for (int nt = 0; nt < 4; ++nt) {
            Frag16 b;
            const bf16_t* src = wf + ((size_t)(kt * 4 + nt) * 32 + lane) * 16;
            b.q[0] = *(const uint4*)(src);
            b.q[1] = *(const uint4*)(src + 8);
            acc[nt] = wmma_bf16(a.v, b.v, acc[nt]);
        }
    }

    const float* bias = (p == 0) ? bq : (p == 1) ? bk : bv;
    bf16_t* dst = (p == 0) ? Q : (p == 1) ? K : V;
    float scale = (p == 0) ? 8.0f : 1.0f;
#pragma unroll
    for (int nt = 0; nt < 4; ++nt) {
        float bn = bias[nt * 16 + n];
#pragma unroll
        for (int r = 0; r < 8; ++r) {       // C-layout: row = r + 8*half
            int row = rt * 16 + r + 8 * h;
            float vv = (acc[nt][r] + bn) * scale;
            dst[(size_t)row * HDIM + nt * 16 + n] = (bf16_t)vv;
        }
    }
}

// ---------------------------------------------------------------------------
// Kernel 3: causal flash attention.  Block = 4 waves = 64 queries of a batch.
// K chunks (32 keys x 64 dims bf16) staged by the Tensor Data Mover into
// double-buffered LDS with native padding (32dw rows + 4dw pad = 36dw stride);
// V staged manually (key-pair swizzle).  All matmuls bf16 WMMA, fp32 softmax.
// ---------------------------------------------------------------------------
#define KS_STRIDE 36                 // dwords per staged K row
#define VP_STRIDE 20                 // dwords per paired-V column
#define P_STRIDE  40                 // bf16 per P row
#define KSZ       (32 * KS_STRIDE * 4)           // 4608 B, one K buffer
#define VPSZ      (HDIM * VP_STRIDE * 4)         // 5120 B, one V buffer
#define VP_BASE   (2 * KSZ)                      // 9216
#define PB_BASE   (VP_BASE + 2 * VPSZ)           // 19456
#define PB_WSZ    (16 * P_STRIDE * 2)            // 1280 B per wave
#define SMEM_BYTES (PB_BASE + 4 * PB_WSZ)        // 24576 B dynamic LDS

#if HAVE_TDM
// Issue one TDM 2-D tile load: 32 rows x 128B from row-major global K into
// LDS at lds_off, padding 4 dwords after every 32 dwords (row).
__device__ __forceinline__ void tdm_load_k(uint64_t gaddr, uint32_t lds_off) {
    u32x4 g0;
    g0[0] = 1u;                                   // count=1, user descriptor
    g0[1] = lds_off;                              // D#.lds_addr (bytes)
    g0[2] = (uint32_t)gaddr;                      // D#.global_addr[31:0]
    g0[3] = (uint32_t)(gaddr >> 32) | (2u << 30); // addr[56:32] | type=2
    i32x8 g1;
    g1[0] = (1 << 16)        // data_size = 2 bytes
          | (1 << 20)        // pad_enable
          | (4 << 22)        // pad_interval: every 32 dwords
          | (3 << 25);       // pad_amount: 4 dwords
    g1[1] = (HDIM & 0xFFFF) << 16;        // tensor_dim0 = 64 (lo16)
    g1[2] = (SEQ << 16);                  // tensor_dim0 hi | tensor_dim1 = 2048
    g1[3] = (HDIM << 16);                 // tensor_dim1 hi | tile_dim0 = 64
    g1[4] = 32;                           // tile_dim1 = 32, tile_dim2 = 0
    g1[5] = HDIM;                         // tensor_dim0_stride = 64 elements
    g1[6] = 0;
    g1[7] = 0;
    i32x4 z4 = {0, 0, 0, 0};
#if TDM_SIX_ARG
    i32x8 z8 = {0, 0, 0, 0, 0, 0, 0, 0};
    __builtin_amdgcn_tensor_load_to_lds(g0, g1, z4, z4, z8, 0);
#else
    __builtin_amdgcn_tensor_load_to_lds(g0, g1, z4, z4, 0);
#endif
}
#endif

__global__ void __launch_bounds__(128)
attn(const bf16_t* __restrict__ Q, const bf16_t* __restrict__ K,
     const bf16_t* __restrict__ V, float* __restrict__ out) {
    extern __shared__ __align__(16) char smem[];   // dynamic only -> offset 0

    int wave = threadIdx.x >> 5;
    int lane = threadIdx.x & 31;
    int n = lane & 15;
    int h = lane >> 4;
    int kb = h * 16;

    int b     = blockIdx.x >> 5;        // batch
    int qt    = blockIdx.x & 31;        // 64-query tile within batch
    int qbase = qt * 64;
    int qbW   = qbase + wave * 16;
    size_t rowbase = (size_t)b * SEQ;

    // Q A-fragments (two 32-wide K steps over head dim)
    const bf16_t* qp = Q + (rowbase + qbW + n) * HDIM;
    Frag16 a0, a1;
    a0.q[0] = *(const uint4*)(qp + kb);
    a0.q[1] = *(const uint4*)(qp + kb + 8);
    a1.q[0] = *(const uint4*)(qp + 32 + kb);
    a1.q[1] = *(const uint4*)(qp + 32 + kb + 8);

    v8f o[4];
#pragma unroll
    for (int t = 0; t < 4; ++t) o[t] = zero8();
    float M[8], L[8];
#pragma unroll
    for (int r = 0; r < 8; ++r) { M[r] = -INFINITY; L[r] = 0.f; }

    const bf16_t*   Kgp  = K + rowbase * HDIM;
    const uint32_t* Kg32 = (const uint32_t*)Kgp;
    const bf16_t*   Vg   = V + rowbase * HDIM;
    int nchunks = qt * 2 + 2;           // keys needed: [0, qbase+64)

#if HAVE_TDM
    if (wave == 0)                       // prologue: chunk 0 -> K buffer 0
        tdm_load_k((uint64_t)(uintptr_t)Kgp, 0u);
#endif

    for (int j = 0; j < nchunks; ++j) {
        int buf   = j & 1;
        int kbase = j * 32;
        const uint32_t* Ks = (const uint32_t*)(smem + buf * KSZ);
        uint32_t*       Vp = (uint32_t*)(smem + VP_BASE + buf * VPSZ);

        // stage V chunk key-paired: Vp[dim][k2] = {V[2k2][dim], V[2k2+1][dim]}
        for (int i = threadIdx.x; i < 1024; i += 128) {
            int k2 = i & 15, nn = i >> 4;
            bf16_t lo = Vg[(size_t)(kbase + 2 * k2) * HDIM + nn];
            bf16_t hi = Vg[(size_t)(kbase + 2 * k2 + 1) * HDIM + nn];
            Vp[nn * VP_STRIDE + k2] = pack_bf16(lo, hi);
        }
        if (j + 1 < nchunks)
            __builtin_prefetch(&Vg[(size_t)(kbase + 32) * HDIM], 0, 0);

#if HAVE_TDM
        if (wave == 0) {                 // pipeline: issue j+1, ensure j done
            if (j + 1 < nchunks) {
                tdm_load_k((uint64_t)(uintptr_t)(Kgp + (size_t)(kbase + 32) * HDIM),
                           (uint32_t)((buf ^ 1) * KSZ));
                __builtin_amdgcn_s_wait_tensorcnt(1);
            } else {
                __builtin_amdgcn_s_wait_tensorcnt(0);
            }
        }
#else
        // fallback: manual K staging
        {
            uint32_t* Kw = (uint32_t*)(smem + buf * KSZ);
            for (int i = threadIdx.x; i < 1024; i += 128) {
                int row = i >> 5, c = i & 31;
                Kw[row * KS_STRIDE + c] = Kg32[(size_t)(kbase + row) * 32 + c];
            }
        }
#endif
        __syncthreads();                 // K(j) + V(j) visible to all waves

        if (kbase <= qbW + 15) {
            // ---- S = Q * K^T (16 queries x 32 keys)
            v8f s[2];
#pragma unroll
            for (int st = 0; st < 2; ++st) {
                const uint32_t* kr = Ks + (size_t)(st * 16 + n) * KS_STRIDE + 8 * h;
                Frag16 b0, b1;
                b0.q[0] = *(const uint4*)(kr);       // head dims 0..31
                b0.q[1] = *(const uint4*)(kr + 4);
                b1.q[0] = *(const uint4*)(kr + 16);  // head dims 32..63
                b1.q[1] = *(const uint4*)(kr + 20);
                v8f z = zero8();
                z = wmma_bf16(a0.v, b0.v, z);
                z = wmma_bf16(a1.v, b1.v, z);
                s[st] = z;
            }
            // ---- causal mask
            if (kbase + 31 > qbW) {
#pragma unroll
                for (int st = 0; st < 2; ++st)
#pragma unroll
                    for (int r = 0; r < 8; ++r) {
                        int qq  = qbW + r + 8 * h;
                        int key = kbase + st * 16 + n;
                        if (key > qq) s[st][r] = -INFINITY;
                    }
            }
            // ---- online softmax (row lives in one VGPR across a 16-lane half)
#pragma unroll
            for (int r = 0; r < 8; ++r) {
                float m = fmaxf(s[0][r], s[1][r]);
#pragma unroll
                for (int msk = 1; msk < 16; msk <<= 1)
                    m = fmaxf(m, __shfl_xor(m, msk, 32));
                float Mn = fmaxf(M[r], m);
                float sc = __expf(M[r] - Mn);
                M[r] = Mn;
                float p0 = __expf(s[0][r] - Mn);
                float p1 = __expf(s[1][r] - Mn);
                s[0][r] = p0; s[1][r] = p1;
                float ls = p0 + p1;
#pragma unroll
                for (int msk = 1; msk < 16; msk <<= 1)
                    ls += __shfl_xor(ls, msk, 32);
                L[r] = L[r] * sc + ls;
                o[0][r] *= sc; o[1][r] *= sc; o[2][r] *= sc; o[3][r] *= sc;
            }
            // ---- P: C-layout -> A-layout via per-wave LDS buffer
            bf16_t* pw = (bf16_t*)(smem + PB_BASE + wave * PB_WSZ);
#pragma unroll
            for (int st = 0; st < 2; ++st)
#pragma unroll
                for (int r = 0; r < 8; ++r)
                    pw[(r + 8 * h) * P_STRIDE + st * 16 + n] = (bf16_t)s[st][r];
            Frag16 pa;
            const bf16_t* pr = pw + n * P_STRIDE + kb;
            pa.q[0] = *(const uint4*)(pr);
            pa.q[1] = *(const uint4*)(pr + 8);
            // ---- O += P(16x32) * V(32x64)
#pragma unroll
            for (int nt = 0; nt < 4; ++nt) {
                const uint32_t* vr = (const uint32_t*)(smem + VP_BASE + buf * VPSZ)
                                   + (size_t)(nt * 16 + n) * VP_STRIDE + 8 * h;
                Frag16 bv_;
                bv_.q[0] = *(const uint4*)(vr);
                bv_.q[1] = *(const uint4*)(vr + 4);
                o[nt] = wmma_bf16(pa.v, bv_.v, o[nt]);
            }
        }
        __syncthreads();                 // all reads of buf done before reuse
    }

    // epilogue: normalize and store fp32
#pragma unroll
    for (int r = 0; r < 8; ++r) {
        float inv = 1.0f / L[r];
        size_t row = rowbase + qbW + r + 8 * h;
#pragma unroll
        for (int nt = 0; nt < 4; ++nt)
            out[row * HDIM + nt * 16 + n] = o[nt][r] * inv;
    }
}

// ---------------------------------------------------------------------------
// Launch: ws layout = [wfrag 384KB][Q 1MB][K 1MB][V 1MB]  (~3.4 MB total)
// ---------------------------------------------------------------------------
extern "C" void kernel_launch(void* const* d_in, const int* in_sizes, int n_in,
                              void* d_out, int out_size, void* d_ws, size_t ws_size,
                              hipStream_t stream) {
    const float* x  = (const float*)d_in[0];
    const float* Wq = (const float*)d_in[1];
    const float* bq = (const float*)d_in[2];
    const float* Wk = (const float*)d_in[3];
    const float* bk = (const float*)d_in[4];
    const float* Wv = (const float*)d_in[5];
    const float* bv = (const float*)d_in[6];

    char* ws = (char*)d_ws;
    bf16_t* wfrag = (bf16_t*)ws;                       // 3*32*4*32*16*2 = 393216 B
    bf16_t* Qb = (bf16_t*)(ws + 393216);
    bf16_t* Kb = Qb + (size_t)NROWS * HDIM;
    bf16_t* Vb = Kb + (size_t)NROWS * HDIM;

    wprep<<<384, 256, 0, stream>>>(Wq, Wk, Wv, wfrag);
    qkv_proj<<<192, 256, 0, stream>>>(x, wfrag, bq, bk, bv, Qb, Kb, Vb);
    attn<<<NBATCH * 32, 128, SMEM_BYTES, stream>>>(Qb, Kb, Vb, (float*)d_out);
}
